// CompleteAttention_68418829025814
// MI455X (gfx1250) — matrix operand order repaired and checked by reading, not compiled, not run on hardware
//
#include <hip/hip_runtime.h>

typedef __attribute__((ext_vector_type(16))) _Float16 v16h;
typedef __attribute__((ext_vector_type(8)))  float    v8f;

#define NSEQ  3136
#define DIMC  192
#define RLOW  128
#define HEADS 6
#define HD    32
#define BATCH 32

// window_reverse: xm[b][n][:] lives at x[srcrow][:] (rows contiguous in C)
__device__ __forceinline__ long xsrc(int b, int n) {
  int i = n / 56, j = n - i * 56;
  int wi = i / 7, pi = i - wi * 7;
  int wj = j / 7, pj = j - wj * 7;
  return ((long)((b * 64 + wi * 8 + wj) * 49 + pi * 7 + pj)) * DIMC;
}

__device__ __forceinline__ v8f wmma_f16(const v16h& a, const v16h& b, const v8f& c) {
  return __builtin_amdgcn_wmma_f32_16x16x32_f16(false, a, false, b, (short)0, c,
                                                false, false);
}

// A-frag (16x32, M=lane&15): halfs [koff..koff+7] and [koff+16..koff+23], koff=8*(lane>=16)
__device__ __forceinline__ v16h load_afrag(const _Float16* base, int lane) {
  union { uint4 q[2]; v16h v; } u;
  const _Float16* p = base + ((lane >> 4) << 3);
  u.q[0] = *reinterpret_cast<const uint4*>(p);
  u.q[1] = *reinterpret_cast<const uint4*>(p + 16);
  return u.v;
}
// B-frag from N-major storage: halfs [koff..koff+7] and [koff+16..koff+23], koff=8*(lane>=16)
__device__ __forceinline__ v16h load_bfrag(const _Float16* rowbase, int lane) {
  union { uint4 q[2]; v16h v; } u;
  const _Float16* p = rowbase + ((lane >> 4) << 3);
  u.q[0] = *reinterpret_cast<const uint4*>(p);
  u.q[1] = *reinterpret_cast<const uint4*>(p + 16);
  return u.v;
}
// C/D: vgpr r -> M = r + 8*(lane>=16), N = lane&15
__device__ __forceinline__ int cd_m(int lane, int r) { return ((lane >> 4) << 3) + r; }

#define V8F_ZERO {0.f, 0.f, 0.f, 0.f, 0.f, 0.f, 0.f, 0.f}

// ============ Kernel 1: [E;F] -> f16 + row sums (for bias folding) ============
__global__ void k_prep_ef(const float* __restrict__ E_w, const float* __restrict__ F_w,
                          _Float16* __restrict__ EF, float* __restrict__ sEF) {
  int r = blockIdx.x; // 0..255
  const float* src = (r < RLOW) ? (E_w + (long)r * NSEQ) : (F_w + (long)(r - RLOW) * NSEQ);
  float s = 0.f;
  for (int c = threadIdx.x; c < NSEQ; c += blockDim.x) {
    float v = src[c];
    EF[(long)r * NSEQ + c] = (_Float16)v;
    s += v;
  }
  __shared__ float red[256];
  red[threadIdx.x] = s;
  __syncthreads();
  for (int off = 128; off > 0; off >>= 1) {
    if ((int)threadIdx.x < off) red[threadIdx.x] += red[threadIdx.x + off];
    __syncthreads();
  }
  if (threadIdx.x == 0) sEF[r] = red[0];
}

// ==== Kernel 2: qkv_w / proj_w -> f16, keep native [n_out][k_in] layout ======
__global__ void k_prep_w(const float* __restrict__ qkv_w, const float* __restrict__ proj_w,
                         _Float16* __restrict__ wh) {
  long e = (long)blockIdx.x * blockDim.x + threadIdx.x;
  if (e >= 768L * DIMC) return;
  float v = (e < 576L * DIMC) ? qkv_w[e] : proj_w[e - 576L * DIMC];
  wh[e] = (_Float16)v;
}

// == Kernel 3: xT[b][c][n] = window_reverse(x) transposed to channel-major f16 =
__global__ __launch_bounds__(256) void k_prep_xT(const float* __restrict__ x,
                                                 _Float16* __restrict__ xT) {
  const int b  = blockIdx.z;
  const int c0 = blockIdx.y * 32;
  const int n0 = blockIdx.x * 32;
  __shared__ _Float16 tile[32][36];

  {
    int nl = threadIdx.x >> 3;        // 0..31
    int c4 = (threadIdx.x & 7) << 2;  // 0..28
    float4 v = *reinterpret_cast<const float4*>(x + xsrc(b, n0 + nl) + c0 + c4);
    _Float16 h4[4] = {(_Float16)v.x, (_Float16)v.y, (_Float16)v.z, (_Float16)v.w};
    *reinterpret_cast<uint2*>(&tile[nl][c4]) = *reinterpret_cast<uint2*>(h4);
  }
  __syncthreads();
  {
    int cl = threadIdx.x >> 3;        // 0..31
    int n4 = (threadIdx.x & 7) << 2;  // 0..28
    _Float16 o4[4];
    #pragma unroll
    for (int q = 0; q < 4; ++q) o4[q] = tile[n4 + q][cl];
    *reinterpret_cast<uint2*>(xT + ((long)b * DIMC + c0 + cl) * NSEQ + n0 + n4) =
        *reinterpret_cast<uint2*>(o4);
  }
}

// ========= Kernel 4: xlow[b] (256x192) = [E;F] (256x3136) @ xm[b] ============
// 2x2 register blocking: 32x32 output per wave, A/B frags reused across tiles,
// 4 independent WMMA accumulator chains. No LDS.
__global__ __launch_bounds__(128) void k_xlow(const _Float16* __restrict__ EF,
                                              const _Float16* __restrict__ xT,
                                              _Float16* __restrict__ xlow) {
  const int lane = threadIdx.x & 31;
  const int wv   = threadIdx.x >> 5;
  int gw  = blockIdx.x * 4 + wv;  // 1536 waves
  int b   = gw / 48;
  int rem = gw % 48;
  int mt2 = rem / 6, nt2 = rem % 6;

  const _Float16* ar0 = EF + (long)(mt2 * 32 + (lane & 15)) * NSEQ;
  const _Float16* ar1 = ar0 + 16L * NSEQ;
  const _Float16* br0 = xT + ((long)b * DIMC + nt2 * 32 + (lane & 15)) * NSEQ;
  const _Float16* br1 = br0 + 16L * NSEQ;

  v8f acc[2][2] = {{V8F_ZERO, V8F_ZERO}, {V8F_ZERO, V8F_ZERO}};
  for (int kc = 0; kc < NSEQ / 32; ++kc) {
    v16h a0 = load_afrag(ar0 + kc * 32, lane);
    v16h a1 = load_afrag(ar1 + kc * 32, lane);
    v16h b0 = load_bfrag(br0 + kc * 32, lane);
    v16h b1 = load_bfrag(br1 + kc * 32, lane);
    acc[0][0] = wmma_f16(a0, b0, acc[0][0]);
    acc[0][1] = wmma_f16(a0, b1, acc[0][1]);
    acc[1][0] = wmma_f16(a1, b0, acc[1][0]);
    acc[1][1] = wmma_f16(a1, b1, acc[1][1]);
  }
  #pragma unroll
  for (int ti = 0; ti < 2; ++ti)
    #pragma unroll
    for (int tj = 0; tj < 2; ++tj)
      #pragma unroll
      for (int r = 0; r < 8; ++r) {
        int m = mt2 * 32 + ti * 16 + cd_m(lane, r);
        int n = nt2 * 32 + tj * 16 + (lane & 15);
        xlow[((long)b * 256 + m) * DIMC + n] = (_Float16)acc[ti][tj][r];
      }
}

// == Kernel 5: klow/vlow = xlow @ Wk/Wv^T + folded biases; scale into klow ====
// 2x2 register blocking. klow stored [r][d], vlowT stored [d][r] (N-major forms).
__global__ __launch_bounds__(128) void k_lowrank(
    const _Float16* __restrict__ xlow, const _Float16* __restrict__ wh,
    const float* __restrict__ qkv_b, const float* __restrict__ E_b,
    const float* __restrict__ F_b, const float* __restrict__ sEF,
    _Float16* __restrict__ klow, _Float16* __restrict__ vlowT) {
  const int lane = threadIdx.x & 31;
  const int wv   = threadIdx.x >> 5;
  int gw  = blockIdx.x * 4 + wv;  // 1536 waves
  int b   = gw / 48;
  int rem = gw % 48;
  int kv  = rem / 24;  rem %= 24; // 0=k(E), 1=v(F)
  int mt2 = rem / 6, nt2 = rem % 6;

  const _Float16* ar0 = xlow + ((long)b * 256 + kv * RLOW + mt2 * 32 + (lane & 15)) * DIMC;
  const _Float16* ar1 = ar0 + 16L * DIMC;
  const _Float16* br0 = wh + (long)((1 + kv) * DIMC + nt2 * 32 + (lane & 15)) * DIMC;
  const _Float16* br1 = br0 + 16L * DIMC;

  v8f acc[2][2] = {{V8F_ZERO, V8F_ZERO}, {V8F_ZERO, V8F_ZERO}};
  for (int kc = 0; kc < DIMC / 32; ++kc) {
    v16h a0 = load_afrag(ar0 + kc * 32, lane);
    v16h a1 = load_afrag(ar1 + kc * 32, lane);
    v16h b0 = load_bfrag(br0 + kc * 32, lane);
    v16h b1 = load_bfrag(br1 + kc * 32, lane);
    acc[0][0] = wmma_f16(a0, b0, acc[0][0]);
    acc[0][1] = wmma_f16(a0, b1, acc[0][1]);
    acc[1][0] = wmma_f16(a1, b0, acc[1][0]);
    acc[1][1] = wmma_f16(a1, b1, acc[1][1]);
  }

  const float scale = 0.17677669529663689f; // 1/sqrt(32), folded into klow
  #pragma unroll
  for (int ti = 0; ti < 2; ++ti)
    #pragma unroll
    for (int tj = 0; tj < 2; ++tj) {
      int dg = nt2 * 32 + tj * 16 + (lane & 15); // 0..191
      int h = dg >> 5, d = dg & 31;
      #pragma unroll
      for (int r = 0; r < 8; ++r) {
        int rr = mt2 * 32 + ti * 16 + cd_m(lane, r);
        float v = acc[ti][tj][r];
        if (kv == 0) {
          v = (v + sEF[rr] * qkv_b[DIMC + dg] + E_b[rr]) * scale;
          klow[(((long)b * HEADS + h) * RLOW + rr) * HD + d] = (_Float16)v;   // [r][d]
        } else {
          v = v + sEF[RLOW + rr] * qkv_b[2 * DIMC + dg] + F_b[rr];
          vlowT[(((long)b * HEADS + h) * HD + d) * RLOW + rr] = (_Float16)v;  // [d][r]
        }
      }
    }
}

// ===== Kernel 6: fused Q-gemm -> logits -> softmax -> AV  (per b,head,tile) ==
__global__ __launch_bounds__(128) void k_attn(
    const float* __restrict__ x, const _Float16* __restrict__ wh,
    const float* __restrict__ qkv_b, const _Float16* __restrict__ klow,
    const _Float16* __restrict__ vlowT, float* __restrict__ out) {
  const int lane = threadIdx.x & 31;
  const int wv   = threadIdx.x >> 5;
  const int tile = blockIdx.x; // 0..97 (32 rows)
  const int h    = blockIdx.y;
  const int b    = blockIdx.z;

  __shared__ alignas(16) _Float16 xl[32 * DIMC];   // x tile f16 (row-major)
  __shared__ alignas(16) _Float16 ql[32 * HD];     // q tile
  __shared__ alignas(16) _Float16 kl[RLOW * HD];   // klow [r][d] (N-major B)
  __shared__ alignas(16) _Float16 vl[HD * RLOW];   // vlowT [d][r] (N-major B)
  __shared__ alignas(16) float    af[32 * RLOW];   // logits f32
  __shared__ alignas(16) _Float16 ab[32 * RLOW];   // probs f16

  for (int e4 = threadIdx.x; e4 < 32 * DIMC / 4; e4 += 128) {
    int row = e4 / (DIMC / 4), c4 = (e4 % (DIMC / 4)) * 4;
    float4 v = *reinterpret_cast<const float4*>(x + xsrc(b, tile * 32 + row) + c4);
    _Float16 h4[4] = {(_Float16)v.x, (_Float16)v.y, (_Float16)v.z, (_Float16)v.w};
    *reinterpret_cast<uint2*>(&xl[row * DIMC + c4]) = *reinterpret_cast<uint2*>(h4);
  }
  {
    const uint4* sk = reinterpret_cast<const uint4*>(klow  + ((long)b * HEADS + h) * RLOW * HD);
    const uint4* sv = reinterpret_cast<const uint4*>(vlowT + ((long)b * HEADS + h) * HD * RLOW);
    uint4* dk = reinterpret_cast<uint4*>(kl);
    uint4* dv = reinterpret_cast<uint4*>(vl);
    for (int e = threadIdx.x; e < (HD * RLOW) / 8; e += 128) { dk[e] = sk[e]; dv[e] = sv[e]; }
  }
  __syncthreads();

  { // Q = x @ WqT + bq : 2x2 tiles, B straight from global f16 weights
    int mt = wv >> 1, nt = wv & 1;
    const _Float16* arow = xl + (mt * 16 + (lane & 15)) * DIMC;
    const _Float16* brow = wh + (long)(h * HD + nt * 16 + (lane & 15)) * DIMC;
    v8f acc = V8F_ZERO;
    for (int kc = 0; kc < DIMC / 32; ++kc) {
      v16h a  = load_afrag(arow + kc * 32, lane);
      v16h bb = load_bfrag(brow + kc * 32, lane);
      acc = wmma_f16(a, bb, acc);
    }
    float bias = qkv_b[h * HD + nt * 16 + (lane & 15)];
    #pragma unroll
    for (int r = 0; r < 8; ++r)
      ql[(mt * 16 + cd_m(lane, r)) * HD + nt * 16 + (lane & 15)] = (_Float16)(acc[r] + bias);
  }
  __syncthreads();

  // logits = q @ klow^T : 2x8 tiles, K=32 -> one WMMA each
  for (int t = wv; t < 16; t += 4) {
    int mt = t >> 3, nt = t & 7;
    v16h a  = load_afrag(ql + (mt * 16 + (lane & 15)) * HD, lane);
    v16h bb = load_bfrag(kl + (nt * 16 + (lane & 15)) * HD, lane);
    v8f acc = V8F_ZERO;
    acc = wmma_f16(a, bb, acc);
    #pragma unroll
    for (int r = 0; r < 8; ++r)
      af[(mt * 16 + cd_m(lane, r)) * RLOW + nt * 16 + (lane & 15)] = acc[r];
  }
  __syncthreads();

  if (threadIdx.x < 32) { // row softmax over R=128
    float* row = af + threadIdx.x * RLOW;
    float mx = row[0];
    for (int j = 1; j < RLOW; ++j) mx = fmaxf(mx, row[j]);
    float s = 0.f;
    for (int j = 0; j < RLOW; ++j) s += __expf(row[j] - mx);
    float inv = 1.f / s;
    for (int j = 0; j < RLOW; ++j)
      ab[threadIdx.x * RLOW + j] = (_Float16)(__expf(row[j] - mx) * inv);
  }
  __syncthreads();

  { // out = probs @ vlow : 2x2 tiles, K=128
    int mt = wv >> 1, nt = wv & 1;
    const _Float16* arow = ab + (mt * 16 + (lane & 15)) * RLOW;
    const _Float16* brow = vl + (nt * 16 + (lane & 15)) * RLOW;
    v8f acc = V8F_ZERO;
    for (int kc = 0; kc < RLOW / 32; ++kc) {
      v16h a  = load_afrag(arow + kc * 32, lane);
      v16h bb = load_bfrag(brow + kc * 32, lane);
      acc = wmma_f16(a, bb, acc);
    }
    #pragma unroll
    for (int r = 0; r < 8; ++r) {
      int n = tile * 32 + mt * 16 + cd_m(lane, r);
      int c = h * HD + nt * 16 + (lane & 15);
      out[((long)b * NSEQ + n) * DIMC + c] = acc[r];
    }
  }
}

// ======= Kernel 7: in-place row-block proj GEMM: out = out @ projT + b =======
// A-fragments preloaded once per wave and reused across its 6 N-tiles.
__global__ __launch_bounds__(128) void k_proj(const _Float16* __restrict__ wh,
                                              const float* __restrict__ proj_b,
                                              float* __restrict__ out) {
  const int lane = threadIdx.x & 31;
  const int wv   = threadIdx.x >> 5;
  const int tile = blockIdx.x; // 0..97
  const int b    = blockIdx.y;

  __shared__ alignas(16) _Float16 ol[32 * DIMC];
  const long rowbase = ((long)b * NSEQ + tile * 32) * DIMC;

  for (int e4 = threadIdx.x; e4 < 32 * DIMC / 4; e4 += 128) {
    int e = e4 * 4;
    float4 v = *reinterpret_cast<const float4*>(out + rowbase + e);
    _Float16 h4[4] = {(_Float16)v.x, (_Float16)v.y, (_Float16)v.z, (_Float16)v.w};
    *reinterpret_cast<uint2*>(&ol[e]) = *reinterpret_cast<uint2*>(h4);
  }
  __syncthreads(); // all in-place reads complete before any write

  const int mt     = wv >> 1;       // 0..1
  const int ntbase = (wv & 1) * 6;  // 0 or 6
  const _Float16* arow = ol + (mt * 16 + (lane & 15)) * DIMC;
  v16h a[6];
  #pragma unroll
  for (int kc = 0; kc < 6; ++kc) a[kc] = load_afrag(arow + kc * 32, lane);

  const _Float16* ph = wh + 576L * DIMC; // proj weights, [n][k] f16
  for (int j = 0; j < 6; ++j) {
    int nt = ntbase + j;
    const _Float16* brow = ph + (long)(nt * 16 + (lane & 15)) * DIMC;
    v8f acc = V8F_ZERO;
    #pragma unroll
    for (int kc = 0; kc < 6; ++kc)
      acc = wmma_f16(a[kc], load_bfrag(brow + kc * 32, lane), acc);
    float bias = proj_b[nt * 16 + (lane & 15)];
    #pragma unroll
    for (int r = 0; r < 8; ++r)
      out[rowbase + (mt * 16 + cd_m(lane, r)) * DIMC + nt * 16 + (lane & 15)] = acc[r] + bias;
  }
}

// ============================== launcher =====================================
extern "C" void kernel_launch(void* const* d_in, const int* in_sizes, int n_in,
                              void* d_out, int out_size, void* d_ws, size_t ws_size,
                              hipStream_t stream) {
  (void)in_sizes; (void)n_in; (void)out_size; (void)ws_size;
  const float* x      = (const float*)d_in[0];
  const float* qkv_w  = (const float*)d_in[1];
  const float* qkv_b  = (const float*)d_in[2];
  const float* E_w    = (const float*)d_in[3];
  const float* E_b    = (const float*)d_in[4];
  const float* F_w    = (const float*)d_in[5];
  const float* F_b    = (const float*)d_in[6];
  const float* proj_w = (const float*)d_in[7];
  const float* proj_b = (const float*)d_in[8];
  float* out = (float*)d_out;

  char* ws = (char*)d_ws;
  size_t off = 0;
  _Float16* EF    = (_Float16*)(ws + off); off += (size_t)256 * NSEQ * 2;          // EF f16
  _Float16* wh    = (_Float16*)(ws + off); off += (size_t)768 * DIMC * 2;          // weights f16
  float*    sEF   = (float*)(ws + off);    off += 256 * 4;                         // row sums
  _Float16* xlow  = (_Float16*)(ws + off); off += (size_t)BATCH * 256 * DIMC * 2;  // [E;F]@x
  _Float16* klow  = (_Float16*)(ws + off); off += (size_t)BATCH * HEADS * RLOW * HD * 2;
  _Float16* vlowT = (_Float16*)(ws + off); off += (size_t)BATCH * HEADS * HD * RLOW * 2;
  _Float16* xT    = (_Float16*)(ws + off); off += (size_t)BATCH * DIMC * NSEQ * 2; // ch-major x

  k_prep_ef<<<256, 256, 0, stream>>>(E_w, F_w, EF, sEF);
  k_prep_w<<<(768 * DIMC + 255) / 256, 256, 0, stream>>>(qkv_w, proj_w, wh);
  k_prep_xT<<<dim3(98, 6, BATCH), 256, 0, stream>>>(x, xT);
  k_xlow<<<384, 128, 0, stream>>>(EF, xT, xlow);
  k_lowrank<<<384, 128, 0, stream>>>(xlow, wh, qkv_b, E_b, F_b, sEF, klow, vlowT);
  k_attn<<<dim3(98, HEADS, BATCH), 128, 0, stream>>>(x, wh, qkv_b, klow, vlowT, out);
  k_proj<<<dim3(98, BATCH), 128, 0, stream>>>(wh, proj_b, out);
}
